// dgcnn_26353919328932
// MI455X (gfx1250) — matrix operand order repaired and checked by reading, not compiled, not run on hardware
//
#include <hip/hip_runtime.h>

// ---------------------------------------------------------------------------
// DGCNN forward for MI455X (gfx1250, wave32, WMMA).
//   Feature maps stored POINT-MAJOR [B][N][C] so neighbor gathers, KNN tiles
//   and conv0 staging are all contiguous float4 (b128) accesses.
//   GEMMs (edge conv 1x1, conv0) use v_wmma_f32_16x16x32_f16; every v16h
//   fragment is 2x ds_load_b128 from fragment-linear LDS layouts.
//   conv0 is double-buffered: next slab's global loads are issued before the
//   current slab's WMMAs, converted/stored after -> loads hide under WMMA.
//   KNN distance + top-20 stays f32 VALU (ranking precision, serial select).
// ---------------------------------------------------------------------------

#define NPTS   2048
#define BATCH  16
#define KNN    20
#define HRS    320                     // h row stride (channels per point)

typedef __attribute__((ext_vector_type(16))) _Float16 v16h;
typedef __attribute__((ext_vector_type(8)))  _Float16 v8h;
typedef __attribute__((ext_vector_type(4)))  _Float16 v4h;
typedef __attribute__((ext_vector_type(8)))  float    v8f;

__device__ __forceinline__ v16h cat16(v8h lo, v8h hi)
{
    return __builtin_shufflevector(lo, hi, 0, 1, 2, 3, 4, 5, 6, 7,
                                           8, 9, 10, 11, 12, 13, 14, 15);
}

// ---------------------------------------------------------------------------
// x [B][3][N] -> xT [B][N][4]  (channel pad with 0; distances unaffected)
// ---------------------------------------------------------------------------
__global__ void xpose_kernel(const float* __restrict__ x, float* __restrict__ xT)
{
    int i = blockIdx.x * blockDim.x + threadIdx.x;
    if (i >= BATCH * NPTS) return;
    int b = i / NPTS, n = i % NPTS;
    float4 v;
    v.x = x[((size_t)b * 3 + 0) * NPTS + n];
    v.y = x[((size_t)b * 3 + 1) * NPTS + n];
    v.z = x[((size_t)b * 3 + 2) * NPTS + n];
    v.w = 0.0f;
    *(float4*)&xT[(size_t)i * 4] = v;
}

// ---------------------------------------------------------------------------
// KNN: one thread per point; LDS-tiled sweep; top-20 in registers via an
// unrolled conditional-swap insertion network.
// feats: n-major, element (n,c) at feats[b*batchStride + n*RS + c]
// ---------------------------------------------------------------------------
template<int C>
__global__ __launch_bounds__(128)
void knn_kernel(const float* __restrict__ feats, int RS, int batchStride,
                int* __restrict__ idxOut)
{
    constexpr int TM   = 128;
    constexpr int CPAD = (C % 4 == 0) ? (C + 4) : C;   // 16B-mult stride when vec
    __shared__ __align__(16) float tile[TM][CPAD];
    __shared__ float sqm[TM];

    const int n = blockIdx.x * 128 + threadIdx.x;
    const int b = blockIdx.y;
    const float* xb = feats + (size_t)b * batchStride;

    float f[C];
    float sqn = 0.0f;
    if constexpr (C % 4 == 0) {
#pragma unroll
        for (int c4 = 0; c4 < C / 4; ++c4) {
            float4 v = *(const float4*)&xb[(size_t)n * RS + (c4 << 2)];
            f[4 * c4 + 0] = v.x; f[4 * c4 + 1] = v.y;
            f[4 * c4 + 2] = v.z; f[4 * c4 + 3] = v.w;
        }
    } else {
#pragma unroll
        for (int c = 0; c < C; ++c) f[c] = xb[(size_t)n * RS + c];
    }
#pragma unroll
    for (int c = 0; c < C; ++c) sqn += f[c] * f[c];

    float td[KNN]; int ti[KNN];
#pragma unroll
    for (int j = 0; j < KNN; ++j) { td[j] = 3.0e38f; ti[j] = 0; }

    for (int mt = 0; mt < NPTS; mt += TM) {
        // cooperative tile load: contiguous float4 rows (n-major layout)
        if constexpr (C % 4 == 0) {
#pragma unroll
            for (int grp = 0; grp < 4; ++grp) {          // two-phase: batch loads
                float4 tmp[4];
#pragma unroll
                for (int i = 0; i < 4; ++i) {
                    int idx = threadIdx.x + (grp * 4 + i) * 128;
                    int c4 = idx & (C / 4 - 1), m = idx / (C / 4);
                    tmp[i] = *(const float4*)&xb[(size_t)(mt + m) * RS + (c4 << 2)];
                }
#pragma unroll
                for (int i = 0; i < 4; ++i) {
                    int idx = threadIdx.x + (grp * 4 + i) * 128;
                    int c4 = idx & (C / 4 - 1), m = idx / (C / 4);
                    *(float4*)&tile[m][c4 << 2] = tmp[i];
                }
            }
        } else {
            for (int l = threadIdx.x; l < TM * C; l += 128) {
                int m = l / C, c = l % C;
                tile[m][c] = xb[(size_t)(mt + m) * RS + c];
            }
        }
        // prefetch next candidate tile rows (global_prefetch_b8)
        if (mt + TM < NPTS)
            __builtin_prefetch(&xb[(size_t)(mt + TM + threadIdx.x) * RS], 0, 3);
        __syncthreads();
        {
            float s = 0.0f;
#pragma unroll
            for (int c = 0; c < C; ++c) { float v = tile[threadIdx.x][c]; s += v * v; }
            sqm[threadIdx.x] = s;
        }
        __syncthreads();

        for (int m = 0; m < TM; ++m) {
            const int gm = mt + m;
            float dot = 0.0f;
            if constexpr (C % 4 == 0) {
                const float4* trow = (const float4*)&tile[m][0];  // b128 bcast
#pragma unroll
                for (int c4 = 0; c4 < C / 4; ++c4) {
                    float4 t = trow[c4];
                    dot = fmaf(f[4 * c4 + 0], t.x, dot);
                    dot = fmaf(f[4 * c4 + 1], t.y, dot);
                    dot = fmaf(f[4 * c4 + 2], t.z, dot);
                    dot = fmaf(f[4 * c4 + 3], t.w, dot);
                }
            } else {
#pragma unroll
                for (int c = 0; c < C; ++c) dot = fmaf(f[c], tile[m][c], dot);
            }
            float d = sqn + sqm[m] - 2.0f * dot;
            if ((gm != n) && (d < td[KNN - 1])) {
                td[KNN - 1] = d; ti[KNN - 1] = gm;
#pragma unroll
                for (int j = KNN - 1; j > 0; --j) {
                    if (td[j] < td[j - 1]) {
                        float fd = td[j]; td[j] = td[j - 1]; td[j - 1] = fd;
                        int   fi = ti[j]; ti[j] = ti[j - 1]; ti[j - 1] = fi;
                    }
                }
            }
        }
        __syncthreads();
    }

    int* out = idxOut + ((size_t)b * NPTS + n) * KNN;
#pragma unroll
    for (int j = 0; j < KNN; ++j) out[j] = ti[j];
}

// ---------------------------------------------------------------------------
// Edge conv: block = 2 points; columns = 2 pts * 32 slots (k=20 padded to 32
// by duplicating neighbor 19 -> max unchanged).
//   Ecol [col][Kp] f16 column-major  -> B fragment = 2x ds_load_b128
//   Wl   [F][Kp]   f16 row-major     -> A fragment = 2x ds_load_b128
// n-major input: neighbor gathers are float4 (4 channels per load).
// ---------------------------------------------------------------------------
template<int C, int F>
__global__ __launch_bounds__(256)
void edgeconv_kernel(const float* __restrict__ feats, int inRS, int inBS,
                     const int* __restrict__ idxIn,
                     const float* __restrict__ w,      // [F][2C]
                     const float* __restrict__ bias,   // [F]
                     float* __restrict__ outFeats, int outRS, int outBS)
{
    constexpr int K2   = 2 * C;
    constexpr int Kp   = (K2 <= 32) ? 32 : K2;   // pad K to WMMA granularity
    constexpr int KS   = Kp + 8;                 // halves; 16B-multiple stride
    constexpr int NCOL = 64;                     // 2 points * 32 slots

    __shared__ __align__(16) _Float16 Ecol[NCOL][KS];
    __shared__ __align__(16) _Float16 Wl[F][KS];
    __shared__ __align__(16) float cf[2][(C % 4 == 0) ? C : 4]; // central feats
    __shared__ int nbr[2 * KNN];

    const int b   = blockIdx.y;
    const int n0  = blockIdx.x * 2;
    const int tid = threadIdx.x;
    const float* xb = feats + (size_t)b * inBS;

    if (tid < 2 * KNN)
        nbr[tid] = idxIn[((size_t)b * NPTS + n0 + tid / KNN) * KNN + (tid % KNN)];
    if constexpr (C % 4 == 0) {
        if (tid < 2 * (C / 4)) {
            int p = tid / (C / 4), c4 = tid % (C / 4);
            *(float4*)&cf[p][c4 << 2] =
                *(const float4*)&xb[(size_t)(n0 + p) * inRS + (c4 << 2)];
        }
    } else {
        if (tid < 2 * C) {
            int p = tid / C, c = tid % C;
            cf[p][c] = xb[(size_t)(n0 + p) * inRS + c];
        }
    }

    // stage weights
    if constexpr ((K2 % 4 == 0) && (K2 == Kp)) {
        float4 wv[F * (K2 / 4) / 256 > 0 ? F * (K2 / 4) / 256 : 1];
#pragma unroll
        for (int i = 0; i < F * (K2 / 4) / 256; ++i) {
            int idx = tid + i * 256;
            int o = idx / (K2 / 4), k4 = idx % (K2 / 4);
            wv[i] = *(const float4*)&w[(size_t)o * K2 + (k4 << 2)];
        }
#pragma unroll
        for (int i = 0; i < F * (K2 / 4) / 256; ++i) {
            int idx = tid + i * 256;
            int o = idx / (K2 / 4), k4 = idx % (K2 / 4);
            float4 v = wv[i];
            v4h t = { (_Float16)v.x, (_Float16)v.y, (_Float16)v.z, (_Float16)v.w };
            *(v4h*)&Wl[o][k4 << 2] = t;
        }
    } else {
        for (int l = tid; l < F * Kp; l += 256) {
            int o = l / Kp, k = l % Kp;
            Wl[o][k] = (_Float16)((k < K2) ? w[o * K2 + k] : 0.0f);
        }
    }
    __syncthreads();

    // Build edge features, column-major Ecol[col][k], col = p*32 + s
    if constexpr (C % 4 == 0) {
        // central rows k in [0,C): copy cf
#pragma unroll
        for (int i = 0; i < (C / 4) * NCOL / 256; ++i) {
            int idx = tid + i * 256;
            int col = idx % NCOL, k4 = idx / NCOL;
            int p = col >> 5;
            float4 v = *(const float4*)&cf[p][k4 << 2];
            v4h t = { (_Float16)v.x, (_Float16)v.y, (_Float16)v.z, (_Float16)v.w };
            *(v4h*)&Ecol[col][k4 << 2] = t;
        }
        // diff rows k in [C,2C): float4 neighbor gather (two-phase)
        constexpr int GU = (C / 4) * NCOL / 256;
        float4 gv[GU];
#pragma unroll
        for (int i = 0; i < GU; ++i) {
            int idx = tid + i * 256;
            int col = idx % NCOL, k4 = idx / NCOL;
            int p = col >> 5, s = col & 31;
            int s2 = (s < KNN) ? s : (KNN - 1);          // pad: duplicate nbr 19
            int nb = nbr[p * KNN + s2];
            gv[i] = *(const float4*)&xb[(size_t)nb * inRS + (k4 << 2)];
        }
#pragma unroll
        for (int i = 0; i < GU; ++i) {
            int idx = tid + i * 256;
            int col = idx % NCOL, k4 = idx / NCOL;
            int p = col >> 5;
            float4 cv = *(const float4*)&cf[p][k4 << 2];
            float4 v = gv[i];
            v4h t = { (_Float16)(v.x - cv.x), (_Float16)(v.y - cv.y),
                      (_Float16)(v.z - cv.z), (_Float16)(v.w - cv.w) };
            *(v4h*)&Ecol[col][C + (k4 << 2)] = t;
        }
    } else {
        for (int l = tid; l < Kp * NCOL; l += 256) {
            int k = l / NCOL, col = l % NCOL;
            int p = col >> 5, s = col & 31;
            float v = 0.0f;
            if (k < K2) {
                int c = (k < C) ? k : (k - C);
                float central = cf[p][c];
                if (k < C) {
                    v = central;
                } else {
                    int s2 = (s < KNN) ? s : (KNN - 1);
                    v = xb[(size_t)nbr[p * KNN + s2] * inRS + c] - central;
                }
            }
            Ecol[col][k] = (_Float16)v;
        }
    }
    __syncthreads();

    const int wave = tid >> 5;
    const int lane = tid & 31;
    const int lh   = lane >> 4;      // lane half (0/1)
    const int lm   = lane & 15;
    constexpr int MT    = F / 16;
    constexpr int UNITS = MT * 2;    // Mtiles x points (uniform per-wave count)
    constexpr int NKS   = Kp / 32;

#pragma unroll
    for (int uu = 0; uu < UNITS / 8; ++uu) {
        const int u  = wave + uu * 8;
        const int mt = u % MT;
        const int p  = u / MT;

        const _Float16* wrow = &Wl[mt * 16 + lm][0];
        const _Float16* e0   = &Ecol[(p << 5) + lm][0];
        const _Float16* e1   = &Ecol[(p << 5) + 16 + lm][0];

        v8f acc0, acc1;
#pragma unroll
        for (int r = 0; r < 8; ++r) { acc0[r] = 0.0f; acc1[r] = 0.0f; }

        v16h a  = cat16(*(const v8h*)(wrow + lh * 16),
                        *(const v8h*)(wrow + lh * 16 + 8));
        v16h b0 = cat16(*(const v8h*)(e0 + lh * 16),
                        *(const v8h*)(e0 + lh * 16 + 8));
        v16h b1 = cat16(*(const v8h*)(e1 + lh * 16),
                        *(const v8h*)(e1 + lh * 16 + 8));
#pragma unroll
        for (int ks = 0; ks < NKS; ++ks) {
            v16h ca = a, cb0 = b0, cb1 = b1;
            if (ks + 1 < NKS) {
                const int ko = (ks + 1) * 32;
                a  = cat16(*(const v8h*)(wrow + ko + lh * 16),
                           *(const v8h*)(wrow + ko + lh * 16 + 8));
                b0 = cat16(*(const v8h*)(e0 + ko + lh * 16),
                           *(const v8h*)(e0 + ko + lh * 16 + 8));
                b1 = cat16(*(const v8h*)(e1 + ko + lh * 16),
                           *(const v8h*)(e1 + ko + lh * 16 + 8));
            }
            acc0 = __builtin_amdgcn_wmma_f32_16x16x32_f16(false, ca, false, cb0,
                                                          (short)0, acc0, false, false);
            acc1 = __builtin_amdgcn_wmma_f32_16x16x32_f16(false, ca, false, cb1,
                                                          (short)0, acc1, false, false);
        }

        // max over the 32 slots of point p; D layout: M = mt*16 + r + 8*lh
#pragma unroll
        for (int r = 0; r < 8; ++r) {
            float v = fmaxf(acc0[r], acc1[r]);
            v = fmaxf(v, __shfl_xor(v, 1));
            v = fmaxf(v, __shfl_xor(v, 2));
            v = fmaxf(v, __shfl_xor(v, 4));
            v = fmaxf(v, __shfl_xor(v, 8));
            if (lm == 0) {
                int o = mt * 16 + r + (lh << 3);
                outFeats[(size_t)b * outBS + (size_t)(n0 + p) * outRS + o]
                    = v + bias[o];
            }
        }
    }
}

// ---------------------------------------------------------------------------
// conv0 (1024x320 GEMM over 2048 points/batch) + fused bias/affine/ReLU and
// global-max-pool via nonneg-float atomicMax(int).  128x128 tile per block.
// DOUBLE-BUFFERED LDS: per slab, issue next slab's 8 global b128 loads,
// run the 8 WMMAs of the current slab, then cvt_pk+store the staged data
// (s_wait_loadcnt lands after the WMMA block), one barrier per slab.
// ---------------------------------------------------------------------------
__global__ __launch_bounds__(256)
void conv0_kernel(const float* __restrict__ h,      // [B][N][320] n-major
                  const float* __restrict__ w,      // [1024][320]
                  const float* __restrict__ bias,
                  const float* __restrict__ scale,
                  const float* __restrict__ shift,
                  float* __restrict__ g)            // [B][1024] (pre-zeroed)
{
    constexpr int KTOT = 320, MB = 128, NB = 128;
    constexpr int KS = 40;                           // 32+8 halves = 80B stride
    constexpr int NSLAB = KTOT / 32;                 // 10
    __shared__ __align__(16) _Float16 Ws[2][MB][KS];
    __shared__ __align__(16) _Float16 Ht[2][NB][KS]; // col-major: Ht[.][col][k]

    const int b  = blockIdx.z;
    const int n0 = blockIdx.x * NB;
    const int m0 = blockIdx.y * MB;
    const int tid  = threadIdx.x;
    const int wave = tid >> 5, lane = tid & 31, lh = lane >> 4, lm = lane & 15;
    const float* hb = h + (size_t)b * NPTS * HRS;

    const int sm  = tid >> 3, sk4 = tid & 7;         // staging coords (this thread)

    v8f acc[8];
#pragma unroll
    for (int i = 0; i < 8; ++i)
#pragma unroll
        for (int r = 0; r < 8; ++r) acc[i][r] = 0.0f;

    // prologue: stage slab 0 into buffer 0
    {
        float4 wv[4], hv[4];
#pragma unroll
        for (int i = 0; i < 4; ++i)
            wv[i] = *(const float4*)&w[(size_t)(m0 + sm + i * 32) * KTOT + (sk4 << 2)];
#pragma unroll
        for (int i = 0; i < 4; ++i)
            hv[i] = *(const float4*)&hb[(size_t)(n0 + sm + i * 32) * HRS + (sk4 << 2)];
#pragma unroll
        for (int i = 0; i < 4; ++i) {
            float4 v = wv[i];
            v4h t = { (_Float16)v.x, (_Float16)v.y, (_Float16)v.z, (_Float16)v.w };
            *(v4h*)&Ws[0][sm + i * 32][sk4 << 2] = t;
        }
#pragma unroll
        for (int i = 0; i < 4; ++i) {
            float4 v = hv[i];
            v4h t = { (_Float16)v.x, (_Float16)v.y, (_Float16)v.z, (_Float16)v.w };
            *(v4h*)&Ht[0][sm + i * 32][sk4 << 2] = t;
        }
    }
    __syncthreads();

    int cur = 0;
    for (int si = 0; si < NSLAB; ++si) {
        const bool haveNext = (si + 1 < NSLAB);
        float4 wv[4], hv[4];
        // (1) issue next slab's global loads (distinct quads; no wait yet)
        if (haveNext) {
            const int ksn = (si + 1) * 32;
#pragma unroll
            for (int i = 0; i < 4; ++i)
                wv[i] = *(const float4*)&w[(size_t)(m0 + sm + i * 32) * KTOT + ksn + (sk4 << 2)];
#pragma unroll
            for (int i = 0; i < 4; ++i)
                hv[i] = *(const float4*)&hb[(size_t)(n0 + sm + i * 32) * HRS + ksn + (sk4 << 2)];
        }

        // (2) WMMAs for the current slab (loads fly underneath)
        v16h a;
        {
            const _Float16* wrow = &Ws[cur][wave * 16 + lm][0];
            a = cat16(*(const v8h*)(wrow + lh * 16),
                      *(const v8h*)(wrow + lh * 16 + 8));
        }
        v16h bf;
        {
            const _Float16* hcol = &Ht[cur][lm][0] + lh * 16;
            bf = cat16(*(const v8h*)hcol, *(const v8h*)(hcol + 8));
        }
#pragma unroll
        for (int nt = 0; nt < 8; ++nt) {
            v16h curb = bf;
            if (nt < 7) {
                const _Float16* hc = &Ht[cur][(nt + 1) * 16 + lm][0] + lh * 16;
                bf = cat16(*(const v8h*)hc, *(const v8h*)(hc + 8));
            }
            acc[nt] = __builtin_amdgcn_wmma_f32_16x16x32_f16(false, a, false, curb,
                                                             (short)0, acc[nt],
                                                             false, false);
        }

        // (3) convert + store staged data into the other buffer
        if (haveNext) {
#pragma unroll
            for (int i = 0; i < 4; ++i) {
                float4 v = wv[i];
                v4h t = { (_Float16)v.x, (_Float16)v.y, (_Float16)v.z, (_Float16)v.w };
                *(v4h*)&Ws[1 - cur][sm + i * 32][sk4 << 2] = t;
            }
#pragma unroll
            for (int i = 0; i < 4; ++i) {
                float4 v = hv[i];
                v4h t = { (_Float16)v.x, (_Float16)v.y, (_Float16)v.z, (_Float16)v.w };
                *(v4h*)&Ht[1 - cur][sm + i * 32][sk4 << 2] = t;
            }
        }
        // (4) one barrier per slab
        __syncthreads();
        cur ^= 1;
    }

    // epilogue: bias -> affine -> ReLU -> max over 128 cols -> atomic max
#pragma unroll
    for (int r = 0; r < 8; ++r) {
        const int o = m0 + wave * 16 + r + (lh << 3);
        const float bo = bias[o], so = scale[o], to = shift[o];
        float vmax = 0.0f;
#pragma unroll
        for (int nt = 0; nt < 8; ++nt) {
            float y = acc[nt][r] + bo;
            y = fmaxf(y * so + to, 0.0f);
            vmax = fmaxf(vmax, y);
        }
        vmax = fmaxf(vmax, __shfl_xor(vmax, 1));
        vmax = fmaxf(vmax, __shfl_xor(vmax, 2));
        vmax = fmaxf(vmax, __shfl_xor(vmax, 4));
        vmax = fmaxf(vmax, __shfl_xor(vmax, 8));
        if (lm == 0)
            atomicMax((int*)&g[b * 1024 + o], __float_as_int(vmax)); // vmax >= 0
    }
}

// ---------------------------------------------------------------------------
// tiny FC: out = relu((in @ w^T + b) * s + t)
// ---------------------------------------------------------------------------
__global__ void fc_kernel(const float* __restrict__ in, const float* __restrict__ w,
                          const float* __restrict__ bias, const float* __restrict__ scale,
                          const float* __restrict__ shift, float* __restrict__ out,
                          int Cin, int Cout)
{
    int i = blockIdx.x * blockDim.x + threadIdx.x;
    if (i >= BATCH * Cout) return;
    int b = i / Cout, o = i % Cout;
    const float* ib = in + (size_t)b * Cin;
    const float* wr = w + (size_t)o * Cin;
    float acc = bias[o];
    for (int c = 0; c < Cin; ++c) acc = fmaf(ib[c], wr[c], acc);
    out[i] = fmaxf(acc * scale[o] + shift[o], 0.0f);
}

__global__ void zero_kernel(float* p, int n)
{
    int i = blockIdx.x * blockDim.x + threadIdx.x;
    if (i < n) p[i] = 0.0f;
}

// ---------------------------------------------------------------------------
extern "C" void kernel_launch(void* const* d_in, const int* in_sizes, int n_in,
                              void* d_out, int out_size, void* d_ws, size_t ws_size,
                              hipStream_t stream)
{
    const float* x       = (const float*)d_in[0];
    const float* ec0_w   = (const float*)d_in[1];
    const float* ec0_b   = (const float*)d_in[2];
    const float* ec1_w   = (const float*)d_in[3];
    const float* ec1_b   = (const float*)d_in[4];
    const float* ec2_w   = (const float*)d_in[5];
    const float* ec2_b   = (const float*)d_in[6];
    const float* ec3_w   = (const float*)d_in[7];
    const float* ec3_b   = (const float*)d_in[8];
    const float* conv0_w = (const float*)d_in[9];
    const float* conv0_b = (const float*)d_in[10];
    const float* conv0_s = (const float*)d_in[11];
    const float* conv0_t = (const float*)d_in[12];
    const float* fc0_w   = (const float*)d_in[13];
    const float* fc0_b   = (const float*)d_in[14];
    const float* fc0_s   = (const float*)d_in[15];
    const float* fc0_t   = (const float*)d_in[16];
    const float* fc1_w   = (const float*)d_in[17];
    const float* fc1_b   = (const float*)d_in[18];
    const float* fc1_s   = (const float*)d_in[19];
    const float* fc1_t   = (const float*)d_in[20];
    const float* fc2_w   = (const float*)d_in[21];
    const float* fc2_b   = (const float*)d_in[22];
    const float* fc2_s   = (const float*)d_in[23];
    const float* fc2_t   = (const float*)d_in[24];

    // workspace layout (all n-major)
    float* xT = (float*)d_ws;                              // [B][N][4]   0.5 MB
    float* h  = xT + (size_t)BATCH * NPTS * 4;             // [B][N][320] 40 MB
    size_t hElems = (size_t)BATCH * NPTS * HRS;
    int*   idx = (int*)(h + hElems);                       // [B][N][20]  2.5 MB
    float* g   = (float*)(idx + (size_t)BATCH * NPTS * KNN);
    float* g1  = g + BATCH * 1024;
    float* g2  = g1 + BATCH * 512;

    const int xBS = NPTS * 4;                              // xT batch stride
    const int hBS = NPTS * HRS;                            // h batch stride
    dim3 knnGrid(NPTS / 128, BATCH);
    dim3 ecGrid(NPTS / 2, BATCH);

    xpose_kernel<<<(BATCH * NPTS + 255) / 256, 256, 0, stream>>>(x, xT);

    // edge conv 0: xT (3ch, RS=4) -> h[:, 0:64)
    knn_kernel<3><<<knnGrid, 128, 0, stream>>>(xT, 4, xBS, idx);
    edgeconv_kernel<3, 64><<<ecGrid, 256, 0, stream>>>(
        xT, 4, xBS, idx, ec0_w, ec0_b, h + 0, HRS, hBS);

    // edge conv 1: h[:,0:64) -> h[:,64:128)
    knn_kernel<64><<<knnGrid, 128, 0, stream>>>(h + 0, HRS, hBS, idx);
    edgeconv_kernel<64, 64><<<ecGrid, 256, 0, stream>>>(
        h + 0, HRS, hBS, idx, ec1_w, ec1_b, h + 64, HRS, hBS);

    // edge conv 2: h[:,64:128) -> h[:,128:192)
    knn_kernel<64><<<knnGrid, 128, 0, stream>>>(h + 64, HRS, hBS, idx);
    edgeconv_kernel<64, 64><<<ecGrid, 256, 0, stream>>>(
        h + 64, HRS, hBS, idx, ec2_w, ec2_b, h + 128, HRS, hBS);

    // edge conv 3: h[:,128:192) -> h[:,192:320)
    knn_kernel<64><<<knnGrid, 128, 0, stream>>>(h + 128, HRS, hBS, idx);
    edgeconv_kernel<64, 128><<<ecGrid, 256, 0, stream>>>(
        h + 128, HRS, hBS, idx, ec3_w, ec3_b, h + 192, HRS, hBS);

    // conv0 + BN/ReLU + global max pool
    zero_kernel<<<(BATCH * 1024 + 255) / 256, 256, 0, stream>>>(g, BATCH * 1024);
    dim3 c0Grid(NPTS / 128, 1024 / 128, BATCH);
    conv0_kernel<<<c0Grid, 256, 0, stream>>>(h, conv0_w, conv0_b, conv0_s, conv0_t, g);

    // FC head
    fc_kernel<<<(BATCH * 512 + 255) / 256, 256, 0, stream>>>(g,  fc0_w, fc0_b, fc0_s, fc0_t, g1, 1024, 512);
    fc_kernel<<<(BATCH * 256 + 255) / 256, 256, 0, stream>>>(g1, fc1_w, fc1_b, fc1_s, fc1_t, g2, 512, 256);
    fc_kernel<<<(BATCH * 40  + 255) / 256, 256, 0, stream>>>(g2, fc2_w, fc2_b, fc2_s, fc2_t, (float*)d_out, 256, 40);
}
